// TensorNetwork_22497038696717
// MI455X (gfx1250) — compile-verified, matrix operand order checked by reading, not run on hardware
//
#include <hip/hip_runtime.h>
#include <hip/hip_bf16.h>

// ---------------------------------------------------------------------------
// TensorNet (TorchMD-Net) forward for MI455X / gfx1250.
// GEMMs run through v_wmma_f32_16x16x32_f16 (f16 in, f32 accumulate), one
// wave32 per 16x64 output tile (4 accumulators, A-fragment reused 4x).
// B matrices are stored pre-swizzled in WMMA fragment order so each B
// fragment is a single contiguous 32B-per-lane load (2x global_load_b128)
// instead of 16 strided u16 loads. Scatter/segment-sum via float atomics.
// ---------------------------------------------------------------------------

typedef __attribute__((ext_vector_type(16))) _Float16 v16h;
typedef __attribute__((ext_vector_type(8)))  float    v8f;
typedef _Float16 half_t;

#define PI_F 3.14159265358979f

// ---------------- WMMA fragment helpers (ISA 7.12.2 layouts, wave32) -------

// A fragment: 16x32 f16 tile, row-major source, lda = row stride.
// lanes 0-15 hold M=lane, K=0..7 & 16..23 ; lanes 16-31 hold M=lane-16, K+8.
__device__ __forceinline__ v16h load_fragA(const half_t* __restrict__ A,
                                           int lda, int row0, int k0) {
  int l  = threadIdx.x & 31;
  int m  = l & 15;
  int kh = (l >> 4) << 3;                 // 0 or 8
  const half_t* p = A + (size_t)(row0 + m) * (size_t)lda + k0;
  v16h f;
#pragma unroll
  for (int v = 0; v < 8; ++v) {
    int kb = ((v < 4) ? (2 * v) : (16 + 2 * (v - 4))) + kh;
    f[2 * v]     = p[kb];
    f[2 * v + 1] = p[kb + 1];
  }
  return f;
}

// Swizzled-B storage: tile (kt,nt) of 32x16 holds 32 lanes x 16 halfs,
// exactly the per-lane WMMA B fragment. One v16h load per fragment.
__device__ __forceinline__ v16h load_fragB_sw(const half_t* __restrict__ Bsw,
                                              int ntile16, int kt, int nt) {
  int l = threadIdx.x & 31;
  const half_t* p = Bsw + ((((size_t)kt * ntile16 + nt) * 32) + l) * 16;
  return *(const v16h*)p;
}

// inverse map: element (k, col) of logical B[K,N] -> swizzled offset
__device__ __forceinline__ size_t bsw_index(int k, int col, int ntile16) {
  int kt = k >> 5, ko = k & 31;
  int nt = col >> 4, no = col & 15;
  int lh = (ko >> 3) & 1;
  int b  = ko - (lh << 3);          // 0..7 or 16..23
  int idx = (b < 8) ? b : (b - 8);
  int lane = no + (lh << 4);
  return (((size_t)kt * ntile16 + nt) * 32 + (size_t)lane) * 16 + idx;
}

// ---------------- Generic fused GEMM ---------------------------------------
// C[M,N] = epilogue( A[M,K](f16,row-major) x Bsw[K,N](f16,swizzled) )
// MODE bits: 1=+bias[col]  2=SiLU  4=*rowscale[row]  8=*zij[row*128+(col&127)]
//            16=store f32  32=store f16
// One wave per 16x64 tile; requires M%16==0, N%64==0, K%32==0.
template <int MODE>
__global__ void k_gemm(const half_t* __restrict__ A, const half_t* __restrict__ B,
                       int M, int N, int K,
                       const float* __restrict__ bias,
                       const float* __restrict__ rowscale,
                       const float* __restrict__ zij,
                       float* __restrict__ Cf, half_t* __restrict__ Ch) {
  constexpr bool HB = (MODE & 1) != 0;
  constexpr bool HS = (MODE & 2) != 0;
  constexpr bool HR = (MODE & 4) != 0;
  constexpr bool HZ = (MODE & 8) != 0;
  constexpr bool HF = (MODE & 16) != 0;
  constexpr bool HH = (MODE & 32) != 0;

  int ntile16 = N >> 4;
  int ntile   = N >> 6;
  int tiles = (M >> 4) * ntile;
  int t = blockIdx.x * (blockDim.x >> 5) + (threadIdx.x >> 5);
  if (t >= tiles) return;
  int tm = t / ntile, tn = t - tm * ntile;
  int row0 = tm << 4, col0 = tn << 6;
  int nt0 = tn << 2;

  v8f acc[4] = {{}, {}, {}, {}};
  for (int k0 = 0; k0 < K; k0 += 32) {
    v16h a = load_fragA(A, K, row0, k0);
    int kt = k0 >> 5;
#pragma unroll
    for (int q = 0; q < 4; ++q) {
      v16h b = load_fragB_sw(B, ntile16, kt, nt0 + q);
      acc[q] = __builtin_amdgcn_wmma_f32_16x16x32_f16(false, a, false, b,
                                                      (short)0, acc[q], false, false);
    }
  }

  int l  = threadIdx.x & 31;
  int mb = (l >> 4) << 3;           // C layout: VGPR r -> M=r or M=8+r
#pragma unroll
  for (int q = 0; q < 4; ++q) {
    int col = col0 + q * 16 + (l & 15);
    float bv = HB ? bias[col] : 0.0f;
#pragma unroll
    for (int r = 0; r < 8; ++r) {
      int row = row0 + mb + r;
      float v = acc[q][r] + bv;
      if (HS) v = v * __builtin_amdgcn_rcpf(1.0f + __expf(-v));   // fast SiLU
      if (HR) v *= rowscale[row];
      if (HZ) v *= zij[(size_t)row * 128 + (col & 127)];
      size_t o = (size_t)row * (size_t)N + col;
      if (HF) Cf[o] = v;
      if (HH) Ch[o] = (half_t)v;
    }
  }
}

// ---------------- small utility kernels ------------------------------------

__global__ void k_zerof(float* d, long n) {
  long i = (long)blockIdx.x * blockDim.x + threadIdx.x;
  if (i < n) d[i] = 0.0f;
}

__global__ void k_copyf(const float* __restrict__ s, float* __restrict__ d, int n) {
  int i = blockIdx.x * blockDim.x + threadIdx.x;
  if (i < n) d[i] = s[i];
}

// pack f32 weight (rows x cols) into f16 row-major (A matrices / tables)
__global__ void k_pack_h(const float* __restrict__ src, half_t* __restrict__ dst,
                         int rows, int cols, int dst_ld, int coff) {
  int i = blockIdx.x * blockDim.x + threadIdx.x;
  if (i >= rows * cols) return;
  int r = i / cols, c = i - r * cols;
  dst[(size_t)r * dst_ld + coff + c] = (half_t)src[(size_t)r * cols + c];
}

// pack weight W[NO,KI] (row-major f32) as swizzled B: B[k][coff+n] = W[n][k]
__global__ void k_pack_bsw(const float* __restrict__ W, half_t* __restrict__ Bsw,
                           int NO, int KI, int ntile16, int coff) {
  int i = blockIdx.x * blockDim.x + threadIdx.x;
  if (i >= NO * KI) return;
  int n = i / KI, k = i - n * KI;
  Bsw[bsw_index(k, coff + n, ntile16)] = (half_t)W[i];
}

// ---------------- edge preprocessing ---------------------------------------

__global__ void k_prep_edges(const float* __restrict__ ew, const float* __restrict__ ev,
                             const float* __restrict__ ea, const int* __restrict__ ei,
                             int E, float* __restrict__ cutf, float* __restrict__ vhat,
                             half_t* __restrict__ ea16) {
  int e = blockIdx.x * blockDim.x + threadIdx.x;
  if (e >= E) return;
  float w = ew[e];
  cutf[e] = (w < 5.0f) ? 0.5f * (__cosf(w * (PI_F / 5.0f)) + 1.0f) : 0.0f;
  int s = ei[e], d = ei[E + e];
  float x = ev[e * 3], y = ev[e * 3 + 1], z = ev[e * 3 + 2];
  if (s != d) {
    float n = sqrtf(x * x + y * y + z * z);
    float inv = (n == 0.0f) ? 1.0f : 1.0f / n;
    x *= inv; y *= inv; z *= inv;
  }
  vhat[e * 3] = x; vhat[e * 3 + 1] = y; vhat[e * 3 + 2] = z;
#pragma unroll
  for (int j = 0; j < 32; ++j) ea16[(size_t)e * 32 + j] = (half_t)ea[(size_t)e * 32 + j];
}

// zcat[e] = concat(emb[z[dst]], emb[z[src]]) in f16
__global__ void k_gather_zcat(const int* __restrict__ ei, const int* __restrict__ z,
                              const half_t* __restrict__ embH, int E,
                              half_t* __restrict__ zcat) {
  int t = blockIdx.x * blockDim.x + threadIdx.x;
  if (t >= E * 128) return;
  int e = t >> 7, c = t & 127;
  int s = ei[e], d = ei[E + e];
  zcat[(size_t)e * 256 + c]       = embH[(size_t)z[d] * 128 + c];
  zcat[(size_t)e * 256 + 128 + c] = embH[(size_t)z[s] * 128 + c];
}

// embedding scatter: X[dst,c] += a*skew(v) + b*sym(v) + d*I   (a,b,d already
// include Zij and cutoff from the fused GEMM epilogue)
__global__ void k_scatter_embed(const float* __restrict__ abd,
                                const float* __restrict__ vhat,
                                const int* __restrict__ ei, int E,
                                float* __restrict__ X) {
  int t = blockIdx.x * blockDim.x + threadIdx.x;
  if (t >= E * 128) return;
  int e = t >> 7, c = t & 127;
  float a = abd[(size_t)e * 384 + c];
  float b = abd[(size_t)e * 384 + 128 + c];
  float d = abd[(size_t)e * 384 + 256 + c];
  float vx = vhat[e * 3], vy = vhat[e * 3 + 1], vz = vhat[e * 3 + 2];
  float tmv = (vx * vx + vy * vy + vz * vz) * (1.0f / 3.0f);
  float m[9];
  m[0] = b * (vx * vx - tmv) + d;
  m[1] = a * (-vz) + b * (vx * vy);
  m[2] = a * ( vy) + b * (vx * vz);
  m[3] = a * ( vz) + b * (vy * vx);
  m[4] = b * (vy * vy - tmv) + d;
  m[5] = a * (-vx) + b * (vy * vz);
  m[6] = a * (-vy) + b * (vz * vx);
  m[7] = a * ( vx) + b * (vz * vy);
  m[8] = b * (vz * vz - tmv) + d;
  float* Xp = X + (size_t)ei[E + e] * 1152 + c * 9;
#pragma unroll
  for (int i = 0; i < 9; ++i) atomicAdd(Xp + i, m[i]);
}

// per-node layernorm over channels of tensor_norm(X); emits f16 row for GEMM
__global__ void k_node_norm(const float* __restrict__ X, const float* __restrict__ g,
                            const float* __restrict__ b, half_t* __restrict__ normH) {
  int n = blockIdx.x, c = threadIdx.x;
  const float* Xp = X + (size_t)n * 1152 + c * 9;
  float s = 0.0f;
#pragma unroll
  for (int e = 0; e < 9; ++e) { float v = Xp[e]; s += v * v; }
  __shared__ float buf[128];
  buf[c] = s; __syncthreads();
  for (int off = 64; off > 0; off >>= 1) { if (c < off) buf[c] += buf[c + off]; __syncthreads(); }
  float mean = buf[0] * (1.0f / 128.0f);
  __syncthreads();
  float dv = s - mean;
  buf[c] = dv * dv; __syncthreads();
  for (int off = 64; off > 0; off >>= 1) { if (c < off) buf[c] += buf[c + off]; __syncthreads(); }
  float var = buf[0] * (1.0f / 128.0f);
  float y = dv * rsqrtf(var + 1e-5f) * g[c] + b[c];
  normH[(size_t)n * 128 + c] = (half_t)y;
}

// decompose T into I/A/S parts, write into swizzled K=384 B-matrix
// (rows c, 128+c, 256+c; cols n*9..n*9+8)
__device__ __forceinline__ void write_parts(half_t* __restrict__ P, int n, int c,
                                            int ntile16, const float* T) {
  const half_t hz = (half_t)0.0f;
  float tm = (T[0] + T[4] + T[8]) * (1.0f / 3.0f);
  int col = n * 9;
  float Iv[9] = {tm, 0.f, 0.f, 0.f, tm, 0.f, 0.f, 0.f, tm};
  float Av[9];
  Av[0] = 0.f; Av[4] = 0.f; Av[8] = 0.f;
  Av[1] = 0.5f * (T[1] - T[3]); Av[3] = -Av[1];
  Av[2] = 0.5f * (T[2] - T[6]); Av[6] = -Av[2];
  Av[5] = 0.5f * (T[5] - T[7]); Av[7] = -Av[5];
  float Sv[9];
  Sv[0] = T[0] - tm; Sv[4] = T[4] - tm; Sv[8] = T[8] - tm;
  Sv[1] = 0.5f * (T[1] + T[3]); Sv[3] = Sv[1];
  Sv[2] = 0.5f * (T[2] + T[6]); Sv[6] = Sv[2];
  Sv[5] = 0.5f * (T[5] + T[7]); Sv[7] = Sv[5];
  (void)hz;
#pragma unroll
  for (int e = 0; e < 9; ++e) {
    P[bsw_index(c,       col + e, ntile16)] = (half_t)Iv[e];
    P[bsw_index(128 + c, col + e, ntile16)] = (half_t)Av[e];
    P[bsw_index(256 + c, col + e, ntile16)] = (half_t)Sv[e];
  }
}

// build parts from node-major X; prenorm==1 also rescales X by 1/(|X|^2+1)
__global__ void k_parts_from_X(float* __restrict__ X, half_t* __restrict__ partsB,
                               int N, int ntile16, int prenorm) {
  int t = blockIdx.x * blockDim.x + threadIdx.x;
  if (t >= N * 128) return;
  int n = t >> 7, c = t & 127;
  float* Xp = X + (size_t)n * 1152 + c * 9;
  float T[9]; float s = 0.0f;
#pragma unroll
  for (int e = 0; e < 9; ++e) { T[e] = Xp[e]; s += T[e] * T[e]; }
  if (prenorm) {
    float inv = 1.0f / (s + 1.0f);
#pragma unroll
    for (int e = 0; e < 9; ++e) { T[e] *= inv; Xp[e] = T[e]; }
  }
  write_parts(partsB, n, c, ntile16, T);
}

// build parts from channel-major tensor (clin GEMM output, row stride ldn)
__global__ void k_parts_from_ct(const float* __restrict__ Yct, half_t* __restrict__ partsB,
                                int N, int ldn, int ntile16) {
  int t = blockIdx.x * blockDim.x + threadIdx.x;
  if (t >= N * 128) return;
  int n = t >> 7, c = t & 127;
  const float* p = Yct + (size_t)c * ldn + (size_t)n * 9;
  float T[9];
#pragma unroll
  for (int e = 0; e < 9; ++e) T[e] = p[e];
  write_parts(partsB, n, c, ntile16, T);
}

__device__ __forceinline__ void radial3(const float* T, float f0, float f1, float f2,
                                        float* O) {
  float tm = (T[0] + T[4] + T[8]) * (1.0f / 3.0f);
  float ca = 0.5f * (f1 + f2), cb = 0.5f * (f1 - f2), ci = (f0 - f2) * tm;
#pragma unroll
  for (int i = 0; i < 3; ++i)
#pragma unroll
    for (int j = 0; j < 3; ++j)
      O[i * 3 + j] = ca * T[i * 3 + j] - cb * T[j * 3 + i] + ((i == j) ? ci : 0.0f);
}

// embedding tail: X = new_radial_tensor(Xct, f)
__global__ void k_radial_embed(const float* __restrict__ Xct, const float* __restrict__ f,
                               float* __restrict__ X, int N, int ldn) {
  int t = blockIdx.x * blockDim.x + threadIdx.x;
  if (t >= N * 128) return;
  int n = t >> 7, c = t & 127;
  const float* p = Xct + (size_t)c * ldn + (size_t)n * 9;
  float T[9], O[9];
#pragma unroll
  for (int e = 0; e < 9; ++e) T[e] = p[e];
  const float* fp = f + (size_t)n * 384 + c * 3;
  radial3(T, fp[0], fp[1], fp[2], O);
  float* Xp = X + (size_t)n * 1152 + c * 9;
#pragma unroll
  for (int e = 0; e < 9; ++e) Xp[e] = O[e];
}

// layer message scatter: msg[dst,c] += new_radial(Y[src,c], ea[e,c])
__global__ void k_scatter_layer(const float* __restrict__ Yt, const float* __restrict__ eaf,
                                const int* __restrict__ ei, int E, int ldn,
                                float* __restrict__ msg) {
  int t = blockIdx.x * blockDim.x + threadIdx.x;
  if (t >= E * 128) return;
  int e = t >> 7, c = t & 127;
  int s = ei[e], d = ei[E + e];
  const float* Tp = Yt + (size_t)c * ldn + (size_t)s * 9;
  float T[9], O[9];
#pragma unroll
  for (int i = 0; i < 9; ++i) T[i] = Tp[i];
  const float* fp = eaf + (size_t)e * 384 + c * 3;
  radial3(T, fp[0], fp[1], fp[2], O);
  float* Mp = msg + (size_t)d * 1152 + c * 9;
#pragma unroll
  for (int i = 0; i < 9; ++i) atomicAdd(Mp + i, O[i]);
}

__device__ __forceinline__ void mm3(const float* A, const float* B, float* C) {
#pragma unroll
  for (int i = 0; i < 3; ++i)
#pragma unroll
    for (int j = 0; j < 3; ++j)
      C[i * 3 + j] = A[i * 3] * B[j] + A[i * 3 + 1] * B[3 + j] + A[i * 3 + 2] * B[6 + j];
}

// AB = msg@Y + Y@msg ; Z = AB/(|AB|^2+1) -> swizzled f16 B-matrix (K=128)
__global__ void k_AB_parts(const float* __restrict__ msg, const float* __restrict__ Yt,
                           half_t* __restrict__ ZB, int N, int ldn, int ntile16) {
  int t = blockIdx.x * blockDim.x + threadIdx.x;
  if (t >= N * 128) return;
  int n = t >> 7, c = t & 127;
  float M[9], Y[9], AB1[9], AB2[9];
  const float* Mp = msg + (size_t)n * 1152 + c * 9;
  const float* Yp = Yt + (size_t)c * ldn + (size_t)n * 9;
#pragma unroll
  for (int i = 0; i < 9; ++i) { M[i] = Mp[i]; Y[i] = Yp[i]; }
  mm3(M, Y, AB1); mm3(Y, M, AB2);
  float s = 0.0f;
#pragma unroll
  for (int i = 0; i < 9; ++i) { AB1[i] += AB2[i]; s += AB1[i] * AB1[i]; }
  float inv = 1.0f / (s + 1.0f);
  int col = n * 9;
#pragma unroll
  for (int i = 0; i < 9; ++i)
    ZB[bsw_index(c, col + i, ntile16)] = (half_t)(AB1[i] * inv);
}

// X += dX + dX@dX  (dX channel-major)
__global__ void k_update_X(const float* __restrict__ dXt, float* __restrict__ X,
                           int N, int ldn) {
  int t = blockIdx.x * blockDim.x + threadIdx.x;
  if (t >= N * 128) return;
  int n = t >> 7, c = t & 127;
  float D[9], D2[9];
  const float* p = dXt + (size_t)c * ldn + (size_t)n * 9;
#pragma unroll
  for (int i = 0; i < 9; ++i) D[i] = p[i];
  mm3(D, D, D2);
  float* Xp = X + (size_t)n * 1152 + c * 9;
#pragma unroll
  for (int i = 0; i < 9; ++i) Xp[i] += D[i] + D2[i];
}

// ---------------------------------------------------------------------------

template <int MODE>
static void gemm_launch(hipStream_t stream, const half_t* A, const half_t* B,
                        int M, int N, int K, const float* bias, const float* rs,
                        const float* zij, float* Cf, half_t* Ch) {
  int tiles = (M >> 4) * (N >> 6);
  hipLaunchKernelGGL((k_gemm<MODE>), dim3((tiles + 3) / 4), dim3(128), 0, stream,
                     A, B, M, N, K, bias, rs, zij, Cf, Ch);
}

// MODE bits: 1=bias 2=silu 4=rowscale 8=zij 16=f32 out 32=f16 out
enum {
  GM_BIAS_F32      = 1 | 16,           // Zij
  GM_ABD           = 1 | 4 | 8 | 16,   // dp GEMM: bias, *cutoff, *Zij
  GM_PLAIN_F32     = 16,               // clin GEMMs
  GM_MLP_F16       = 1 | 2 | 32,       // bias+SiLU -> f16 activations
  GM_MLP_F32       = 1 | 2 | 16,       // bias+SiLU -> f32
  GM_MLP_CUT_F32   = 1 | 2 | 4 | 16    // bias+SiLU, *cutoff -> f32 (ea)
};

extern "C" void kernel_launch(void* const* d_in, const int* in_sizes, int n_in,
                              void* d_out, int out_size, void* d_ws, size_t ws_size,
                              hipStream_t stream) {
  if (n_in < 49) return;
  const int*   z  = (const int*)d_in[0];
  const int*   ei = (const int*)d_in[1];
  const float* ew = (const float*)d_in[2];
  const float* ev = (const float*)d_in[3];
  const float* ea = (const float*)d_in[4];
  const int N = in_sizes[0];
  const int E = in_sizes[2];
  const int NC9  = N * 9;
  const int NC9p = (NC9 + 63) & ~63;    // pad to N%64==0 for guard-free tiles
  const int NTp  = NC9p >> 4;           // col tiles of the node-major GEMMs

  // param leaves (jax pytree order: dict keys sorted alphabetically)
  // emb_block: dp_b0-2(5..7) dp_w0-2(8..10) emb(11) emb2_b(12) emb2_w(13)
  //            ln_b(14) ln_g(15) ls_b0(16) ls_b1(17) ls_w0(18) ls_w1(19) lt0-2(20..22)
  // layer l @ 23+13l: l_b0-2, l_w0-2, lt0-2, ol0-3
#define PF(i) ((const float*)d_in[(i)])

  // ---- workspace carve ----
  char* wsb = (char*)d_ws;
  size_t cur = 0;
  auto carve = [&](size_t bytes) -> void* {
    void* p = (void*)(wsb + cur);
    cur = (cur + bytes + 255) & ~(size_t)255;
    return p;
  };
  half_t* embH   = (half_t*)carve(128 * 128 * 2);
  half_t* emb2wT = (half_t*)carve(256 * 128 * 2);
  half_t* dpwT   = (half_t*)carve(32 * 384 * 2);
  float*  dpb    = (float*)carve(384 * 4);
  half_t* ltcatE = (half_t*)carve(128 * 384 * 2);
  half_t* lsw0T  = (half_t*)carve(128 * 256 * 2);
  half_t* lsw1T  = (half_t*)carve(256 * 384 * 2);
  half_t *lw0T[2], *lw1T[2], *lw2T[2], *olcat[2], *ol3A[2], *ltcatL[2];
  for (int l = 0; l < 2; ++l) {
    lw0T[l]   = (half_t*)carve(32 * 128 * 2);
    lw1T[l]   = (half_t*)carve(128 * 256 * 2);
    lw2T[l]   = (half_t*)carve(256 * 384 * 2);
    olcat[l]  = (half_t*)carve(128 * 384 * 2);
    ol3A[l]   = (half_t*)carve(128 * 128 * 2);
    ltcatL[l] = (half_t*)carve(128 * 384 * 2);
  }
  float*  cutf = (float*)carve((size_t)E * 4);
  float*  vhat = (float*)carve((size_t)E * 3 * 4);
  half_t* ea16 = (half_t*)carve((size_t)E * 32 * 2);
  half_t* bufA = (half_t*)carve((size_t)E * 256 * 2);    // zcat / t2
  float*  bufB = (float*)carve((size_t)E * 128 * 4);     // Zij / t1(f16)
  float*  bufE = (float*)carve((size_t)E * 384 * 4);     // abd / eaf
  half_t* partsB = (half_t*)carve((size_t)384 * NC9p * 2);
  float*  Xct  = (float*)carve((size_t)128 * NC9p * 4);  // clin out / Yt
  float*  Y2t  = (float*)carve((size_t)128 * NC9p * 4);  // Y2 / dX reuse
  half_t* ZB   = (half_t*)carve((size_t)128 * NC9p * 2);
  float*  msg  = (float*)carve((size_t)N * 1152 * 4);
  half_t* normH = (half_t*)carve((size_t)N * 128 * 2);
  half_t* ls1  = (half_t*)carve((size_t)N * 256 * 2);
  float*  fbuf = (float*)carve((size_t)N * 384 * 4);
  (void)ws_size; (void)out_size;

  float* X = (float*)d_out;   // node tensors live in d_out [N,128,3,3]

  // A-matrix / table packs (row-major f16)
  auto packA = [&](const float* src, half_t* dst, int rows, int cols,
                   int dst_ld, int coff) {
    int n = rows * cols;
    hipLaunchKernelGGL(k_pack_h, dim3((n + 255) / 256), dim3(256), 0, stream,
                       src, dst, rows, cols, dst_ld, coff);
  };
  // B-matrix packs (swizzled fragment order); W[NO,KI] -> B[k][coff+n]
  auto packB = [&](const float* src, half_t* dst, int NO, int KI,
                   int ntile16, int coff) {
    int n = NO * KI;
    hipLaunchKernelGGL(k_pack_bsw, dim3((n + 255) / 256), dim3(256), 0, stream,
                       src, dst, NO, KI, ntile16, coff);
  };

  // ---- pack weights to f16 ----
  packA(PF(11), embH, 128, 128, 128, 0);
  packB(PF(13), emb2wT, 128, 256, 128 >> 4, 0);
  for (int i = 0; i < 3; ++i) packB(PF(8 + i), dpwT, 128, 32, 384 >> 4, i * 128);
  for (int i = 0; i < 3; ++i)
    hipLaunchKernelGGL(k_copyf, dim3(1), dim3(128), 0, stream, PF(5 + i), dpb + i * 128, 128);
  for (int i = 0; i < 3; ++i) packA(PF(20 + i), ltcatE, 128, 128, 384, i * 128);
  packB(PF(18), lsw0T, 256, 128, 256 >> 4, 0);
  packB(PF(19), lsw1T, 384, 256, 384 >> 4, 0);
  for (int l = 0; l < 2; ++l) {
    int LB = 23 + 13 * l;
    packB(PF(LB + 3), lw0T[l], 128, 32, 128 >> 4, 0);
    packB(PF(LB + 4), lw1T[l], 256, 128, 256 >> 4, 0);
    packB(PF(LB + 5), lw2T[l], 384, 256, 384 >> 4, 0);
    for (int i = 0; i < 3; ++i) packA(PF(LB + 9 + i), olcat[l], 128, 128, 384, i * 128);
    packA(PF(LB + 12), ol3A[l], 128, 128, 128, 0);
    for (int i = 0; i < 3; ++i) packA(PF(LB + 6 + i), ltcatL[l], 128, 128, 384, i * 128);
  }

  const int EC = E * 128, NCt = N * 128;

  // ---- edge prep ----
  hipLaunchKernelGGL(k_prep_edges, dim3((E + 255) / 256), dim3(256), 0, stream,
                     ew, ev, ea, ei, E, cutf, vhat, ea16);
  hipLaunchKernelGGL(k_gather_zcat, dim3((EC + 255) / 256), dim3(256), 0, stream,
                     ei, z, embH, E, bufA);

  // ---- embedding ----
  gemm_launch<GM_BIAS_F32>(stream, bufA, emb2wT, E, 128, 256, PF(12), nullptr,
                           nullptr, bufB, nullptr);                       // Zij
  gemm_launch<GM_ABD>(stream, ea16, dpwT, E, 384, 32, dpb, cutf, bufB,
                      bufE, nullptr);                                     // a,b,d
  hipLaunchKernelGGL(k_zerof, dim3((unsigned)(((long)N * 1152 + 255) / 256)), dim3(256),
                     0, stream, X, (long)N * 1152);
  hipLaunchKernelGGL(k_scatter_embed, dim3((EC + 255) / 256), dim3(256), 0, stream,
                     bufE, vhat, ei, E, X);
  hipLaunchKernelGGL(k_node_norm, dim3(N), dim3(128), 0, stream, X, PF(15), PF(14), normH);
  hipLaunchKernelGGL(k_parts_from_X, dim3((NCt + 255) / 256), dim3(256), 0, stream,
                     X, partsB, N, NTp, 0);
  gemm_launch<GM_PLAIN_F32>(stream, ltcatE, partsB, 128, NC9p, 384, nullptr, nullptr,
                            nullptr, Xct, nullptr);
  gemm_launch<GM_MLP_F16>(stream, normH, lsw0T, N, 256, 128, PF(16), nullptr,
                          nullptr, nullptr, ls1);
  gemm_launch<GM_MLP_F32>(stream, ls1, lsw1T, N, 384, 256, PF(17), nullptr,
                          nullptr, fbuf, nullptr);
  hipLaunchKernelGGL(k_radial_embed, dim3((NCt + 255) / 256), dim3(256), 0, stream,
                     Xct, fbuf, X, N, NC9p);

  // ---- message layers ----
  for (int l = 0; l < 2; ++l) {
    int LB = 23 + 13 * l;
    half_t* t1 = (half_t*)bufB;
    half_t* t2 = bufA;
    gemm_launch<GM_MLP_F16>(stream, ea16, lw0T[l], E, 128, 32, PF(LB + 0), nullptr,
                            nullptr, nullptr, t1);
    gemm_launch<GM_MLP_F16>(stream, t1, lw1T[l], E, 256, 128, PF(LB + 1), nullptr,
                            nullptr, nullptr, t2);
    gemm_launch<GM_MLP_CUT_F32>(stream, t2, lw2T[l], E, 384, 256, PF(LB + 2), cutf,
                                nullptr, bufE, nullptr);                  // ea
    hipLaunchKernelGGL(k_parts_from_X, dim3((NCt + 255) / 256), dim3(256), 0, stream,
                       X, partsB, N, NTp, 1);                             // prenorm
    gemm_launch<GM_PLAIN_F32>(stream, olcat[l], partsB, 128, NC9p, 384, nullptr, nullptr,
                              nullptr, Xct, nullptr);                     // Y
    hipLaunchKernelGGL(k_zerof, dim3((unsigned)(((long)N * 1152 + 255) / 256)), dim3(256),
                       0, stream, msg, (long)N * 1152);
    hipLaunchKernelGGL(k_scatter_layer, dim3((EC + 255) / 256), dim3(256), 0, stream,
                       Xct, bufE, ei, E, NC9p, msg);
    hipLaunchKernelGGL(k_AB_parts, dim3((NCt + 255) / 256), dim3(256), 0, stream,
                       msg, Xct, ZB, N, NC9p, NTp);
    gemm_launch<GM_PLAIN_F32>(stream, ol3A[l], ZB, 128, NC9p, 128, nullptr, nullptr,
                              nullptr, Y2t, nullptr);
    hipLaunchKernelGGL(k_parts_from_ct, dim3((NCt + 255) / 256), dim3(256), 0, stream,
                       Y2t, partsB, N, NC9p, NTp);
    gemm_launch<GM_PLAIN_F32>(stream, ltcatL[l], partsB, 128, NC9p, 384, nullptr, nullptr,
                              nullptr, Y2t, nullptr);                     // dX
    hipLaunchKernelGGL(k_update_X, dim3((NCt + 255) / 256), dim3(256), 0, stream,
                       Y2t, X, N, NC9p);
  }
#undef PF
}
